// ModulatedChunks_85194971284122
// MI455X (gfx1250) — matrix operand order repaired and checked by reading, not compiled, not run on hardware
//
#include <hip/hip_runtime.h>
#include <hip/hip_bf16.h>

// ---------------------------------------------------------------------------
// Problem constants (from reference)
// ---------------------------------------------------------------------------
constexpr int B_   = 8;
constexpr int T_   = 256;
constexpr int C_   = 1024;
constexpr int QD_  = 768;
constexpr int LQ_  = 24;
constexpr int W_   = 16;
constexpr int K_   = 4;                 // chunks per window
constexpr int NW_  = T_ - W_ + 1;       // 241
constexpr int S_   = 8;
constexpr int H_   = 4;
constexpr int HD_  = C_ / H_;           // 256
constexpr int SL_  = NW_ * K_;          // 964
constexpr int SLP_ = 992;               // SL padded up to a multiple of 32

typedef __attribute__((ext_vector_type(16))) __bf16 v16bf;
typedef __attribute__((ext_vector_type(8)))  float  v8f;
typedef __attribute__((ext_vector_type(4)))  unsigned int u32x4;
typedef __attribute__((ext_vector_type(8)))  int i32x8;
typedef __attribute__((ext_vector_type(4)))  int i32x4;

#define DI __device__ __forceinline__

DI __bf16 f2bf(float f) { return (__bf16)f; }

// ---------------------------------------------------------------------------
// Generic bf16-WMMA GEMM:  C[m,n] = act( sum_k A[m,k] * B(k,n) + bias[n] )
//   A      : (M x K) f32 row-major, leading dim lda
//   B_KN=0 : Bm is (N x K) f32 row-major (weight form, x @ W.T) , ld = ldb
//   B_KN=1 : Bm is (K x N) f32 row-major (matrix form, P @ V)   , ld = ldb
//   Klim   : valid K rows of Bm for B_KN=1 (K loop padded to mult of 32; the
//            TDM zero-fills tile rows past Klim, A pads are zeros)
//
// block = 128 threads (4 waves), block tile 64(M) x 64(N), wave tile 16 x 64.
// The 64x32 B tile is staged into LDS once per K-step by a single
// TENSOR_LOAD_TO_LDS (TDM) issued by wave 0, with the D# pad feature adding
// one DWORD per tile row to kill LDS bank conflicts. WMMA fragment layouts
// per CDNA5 ISA 7.12.2 (wave32).
// ---------------------------------------------------------------------------
template <bool B_KN, int ACT>
__global__ void gemm_wmma_kernel(const float* __restrict__ A,  int lda,
                                 const float* __restrict__ Bm, int ldb,
                                 const float* __restrict__ bias,
                                 float* __restrict__ Cm, int ldc,
                                 int M, int N, int Kd, int Klim)
{
  // padded tile: weight form rows of 32+1 floats (64 rows);
  //              KN form rows of 64+1 floats (32 rows)
  __shared__ float Bs[64 * 33];

  const int lane  = threadIdx.x & 31;
  const int wave  = threadIdx.x >> 5;
  const int col   = lane & 15;
  const int hi    = lane >> 4;
  const int mBase = blockIdx.y * 64 + wave * 16;
  const int nBase = blockIdx.x * 64;

  int mA = mBase + col;
  if (mA > M - 1) mA = M - 1;                 // clamp for edge tiles
  const float* arow = A + (size_t)mA * lda;

  const unsigned lds_base = (unsigned)(uintptr_t)&Bs[0];

  v8f acc[4];
#pragma unroll
  for (int j = 0; j < 4; ++j)
#pragma unroll
    for (int r = 0; r < 8; ++r) acc[j][r] = 0.0f;

  for (int kb = 0; kb < Kd; kb += 32) {
    __syncthreads();                          // previous tile fully consumed
    if (threadIdx.x < 32) {
      // ---- build 2D tensor-DMA descriptor (D#) and fire the TDM ----
      unsigned long long gaddr;
      unsigned d0, d1, t0, t1, padi;
      unsigned long long st = (unsigned long long)ldb;
      if (B_KN) {
        // tile = 32 rows (k) x 64 cols (n), rows past Klim zero-filled
        gaddr = (unsigned long long)(uintptr_t)(Bm + (size_t)kb * ldb + nBase);
        d0 = (unsigned)(N - nBase);  d1 = (unsigned)(Klim - kb);
        t0 = 64; t1 = 32; padi = 5;           // pad every 64 DWORDs
      } else {
        // tile = 64 rows (n) x 32 cols (k), rows past N zero-filled
        gaddr = (unsigned long long)(uintptr_t)(Bm + (size_t)nBase * ldb + kb);
        d0 = (unsigned)(Kd - kb);    d1 = (unsigned)(N - nBase);
        t0 = 32; t1 = 64; padi = 4;           // pad every 32 DWORDs
      }
      u32x4 g0;
      g0.x = 1u;                                            // count=1, user D#
      g0.y = lds_base;                                      // lds_addr
      g0.z = (unsigned)(gaddr & 0xFFFFFFFFull);             // global_addr lo
      g0.w = (unsigned)((gaddr >> 32) & 0x1FFFFFFull)       // global_addr hi
           | 0x80000000u;                                   // type = 2 (image)
      i32x8 g1;
      g1[0] = (int)((2u << 16)                              // data_size = 4B
                  | (1u << 20)                              // pad_enable
                  | (padi << 22));                          // pad_interval
                                                            // pad_amount=0 (1 DW)
      g1[1] = (int)((d0 & 0xFFFFu) << 16);                  // tensor_dim0[15:0]
      g1[2] = (int)(((d0 >> 16) & 0xFFFFu)                  // tensor_dim0[31:16]
                  | ((d1 & 0xFFFFu) << 16));                // tensor_dim1[15:0]
      g1[3] = (int)(((d1 >> 16) & 0xFFFFu)                  // tensor_dim1[31:16]
                  | (t0 << 16));                            // tile_dim0
      g1[4] = (int)(t1 & 0xFFFFu);                          // tile_dim1 (dim2=0)
      g1[5] = (int)(st & 0xFFFFFFFFull);                    // dim0_stride[31:0]
      g1[6] = (int)((st >> 32) & 0xFFFFull);                // dim0_stride[47:32]
      g1[7] = 0;                                            // dim1_stride (2D)
      i32x4 gz = {0, 0, 0, 0};
      i32x8 gz8 = {0, 0, 0, 0, 0, 0, 0, 0};
      __builtin_amdgcn_tensor_load_to_lds(g0, g1, gz, gz, gz8, 0);
      __builtin_amdgcn_s_wait_tensorcnt(0);
    }
    __syncthreads();                          // tile visible to all waves

    // ---- A fragment: 16x32 bf16 from global. lane holds row (lane&15);
    //      e[0..7] <-> K = hi*8 + i, e[8..15] <-> K = 16 + hi*8 + i
    v16bf a;
    {
      const float* p0 = arow + kb + hi * 8;
#pragma unroll
      for (int i = 0; i < 8; ++i) a[i] = f2bf(p0[i]);
#pragma unroll
      for (int i = 0; i < 8; ++i) a[8 + i] = f2bf(p0[16 + i]);
      if (kb + 32 < Kd) __builtin_prefetch(p0 + 32, 0, 1);
    }

#pragma unroll
    for (int j = 0; j < 4; ++j) {
      const int n = j * 16 + col;             // local column inside the tile
      // ---- B fragment: 32x16 bf16 from LDS. lane holds column (lane&15);
      //      e[0..7] <-> K = hi*8 + i, e[8..15] <-> K = 16 + hi*8 + (i-8)
      v16bf b;
      if (B_KN) {
#pragma unroll
        for (int i = 0; i < 16; ++i)
          b[i] = f2bf(Bs[(hi * 8 + i + (i >= 8 ? 8 : 0)) * 65 + n]);
      } else {
#pragma unroll
        for (int i = 0; i < 16; ++i)
          b[i] = f2bf(Bs[n * 33 + hi * 8 + i + (i >= 8 ? 8 : 0)]);
      }
      acc[j] = __builtin_amdgcn_wmma_f32_16x16x32_bf16(
          false, a, false, b, (short)0, acc[j], false, false);
    }
  }

  // ---- store: D layout, VGPR r <-> M = r + 8*(lane>>4), N = lane&15
#pragma unroll
  for (int j = 0; j < 4; ++j) {
    int n = nBase + j * 16 + col;
    if (n >= N) continue;
    float bb = bias ? bias[n] : 0.0f;
#pragma unroll
    for (int r = 0; r < 8; ++r) {
      int m = mBase + r + 8 * hi;
      if (m >= M) continue;
      float v = acc[j][r] + bb;
      if (ACT == 1) v = fmaxf(v, 0.0f);
      Cm[(size_t)m * ldc + n] = v;
    }
  }
}

// ---------------------------------------------------------------------------
// Positional encoding add: y = x + PE(l, c) for (B, L, d) tensors.
// ---------------------------------------------------------------------------
__global__ void pe_add_kernel(const float* __restrict__ x, float* __restrict__ y,
                              int total, int L, int d)
{
  int i = blockIdx.x * blockDim.x + threadIdx.x;
  if (i >= total) return;
  int c = i % d;
  int l = (i / d) % L;
  const float kLog1e4 = 9.210340371976184f;   // ln(10000)
  float freq = __expf((float)(c & ~1) * (-kLog1e4 / (float)d));
  float ang  = (float)l * freq;
  y[i] = x[i] + ((c & 1) ? __cosf(ang) : __sinf(ang));
}

// ---------------------------------------------------------------------------
// labels[b,t] = argmax_l dot(v[b,t,:], e1[b,l,:])   (first max wins)
// one wave per (b,t)
// ---------------------------------------------------------------------------
__global__ void sim_argmax_kernel(const float* __restrict__ v,
                                  const float* __restrict__ e1,
                                  int* __restrict__ labels)
{
  int bt = blockIdx.x;
  int b = bt / T_;
  int lane = threadIdx.x;
  const float* vr = v + (size_t)bt * C_;
  float best = -3.4e38f;
  int bestl = 0;
  for (int l = 0; l < LQ_; ++l) {
    const float* er = e1 + ((size_t)b * LQ_ + l) * C_;
    float acc = 0.f;
    for (int c = lane; c < C_; c += 32) acc += vr[c] * er[c];
#pragma unroll
    for (int m = 16; m > 0; m >>= 1) acc += __shfl_xor(acc, m, 32);
    if (acc > best) { best = acc; bestl = l; }
  }
  if (lane == 0) labels[bt] = bestl;
}

// ---------------------------------------------------------------------------
// mode of 4 labels per (b,nw,k); tie -> smallest label (matches argmax(one_hot))
// ---------------------------------------------------------------------------
__global__ void mode_kernel(const int* __restrict__ labels, int* __restrict__ mode)
{
  int i = blockIdx.x * blockDim.x + threadIdx.x;
  if (i >= B_ * NW_ * K_) return;
  int k  = i % K_;
  int nw = (i / K_) % NW_;
  int b  = i / (K_ * NW_);
  int lab[4];
#pragma unroll
  for (int j = 0; j < 4; ++j) lab[j] = labels[b * T_ + nw + k * 4 + j];
  int bestc = 0, bestl = 1 << 30;
#pragma unroll
  for (int j = 0; j < 4; ++j) {
    int c = 0;
#pragma unroll
    for (int j2 = 0; j2 < 4; ++j2) c += (lab[j2] == lab[j]) ? 1 : 0;
    if (c > bestc || (c == bestc && lab[j] < bestl)) { bestc = c; bestl = lab[j]; }
  }
  mode[i] = bestl;
}

// kk[b,s,:] = e2[b, mode[b,s], :]
__global__ void gather_modk_kernel(const float* __restrict__ e2,
                                   const int* __restrict__ mode,
                                   float* __restrict__ kk)
{
  size_t i = (size_t)blockIdx.x * blockDim.x + threadIdx.x;
  if (i >= (size_t)B_ * SL_ * C_) return;
  int c = (int)(i % C_);
  size_t s = i / C_;                       // flat (b,nw,k)
  int b = (int)(s / (NW_ * K_));
  kk[i] = e2[((size_t)b * LQ_ + mode[s]) * C_ + c];
}

// pooled_pre[b,nw,k,c] = mean_j v[b, nw + 4k + j, c]
__global__ void pool_mean_kernel(const float* __restrict__ v, float* __restrict__ pp)
{
  size_t i = (size_t)blockIdx.x * blockDim.x + threadIdx.x;
  if (i >= (size_t)B_ * SL_ * C_) return;
  int c = (int)(i % C_);
  size_t s = i / C_;
  int k  = (int)(s % K_);
  int nw = (int)((s / K_) % NW_);
  int b  = (int)(s / (K_ * NW_));
  const float* base = v + ((size_t)(b * T_ + nw + k * 4)) * C_ + c;
  pp[i] = 0.25f * (base[0] + base[C_] + base[2 * C_] + base[3 * C_]);
}

// ---------------------------------------------------------------------------
// row softmax over scores (length SL_, row pitch SLP_), scale = 1/sqrt(HD)
// also zeroes the padded tail [SL_, SLP_) so attn@V can run K=SLP_ cleanly.
// ---------------------------------------------------------------------------
__global__ void softmax_kernel(float* __restrict__ sc)
{
  float* row = sc + (size_t)blockIdx.x * SLP_;
  const float scale = 1.0f / 16.0f;         // 1/sqrt(256)
  int tid  = threadIdx.x;
  int lane = tid & 31, wid = tid >> 5;
  __shared__ float sm[8];

  float mx = -3.4e38f;
  for (int i = tid; i < SL_; i += 256) mx = fmaxf(mx, row[i] * scale);
#pragma unroll
  for (int m = 16; m > 0; m >>= 1) mx = fmaxf(mx, __shfl_xor(mx, m, 32));
  if (lane == 0) sm[wid] = mx;
  __syncthreads();
  mx = sm[0];
#pragma unroll
  for (int i = 1; i < 8; ++i) mx = fmaxf(mx, sm[i]);
  __syncthreads();

  float sum = 0.f;
  for (int i = tid; i < SL_; i += 256) {
    float e = __expf(row[i] * scale - mx);
    row[i] = e;
    sum += e;
  }
#pragma unroll
  for (int m = 16; m > 0; m >>= 1) sum += __shfl_xor(sum, m, 32);
  if (lane == 0) sm[wid] = sum;
  __syncthreads();
  sum = 0.f;
#pragma unroll
  for (int i = 0; i < 8; ++i) sum += sm[i];
  float inv = 1.0f / sum;
  for (int i = tid; i < SL_; i += 256) row[i] *= inv;
  for (int i = SL_ + tid; i < SLP_; i += 256) row[i] = 0.0f;
}

// ---------------------------------------------------------------------------
// segment max-pool over tokens, write flat[b, c*S + s]
// ---------------------------------------------------------------------------
__global__ void segpool_kernel(const float* __restrict__ outb,
                               const int* __restrict__ vid_len,
                               float* __restrict__ flat)
{
  int i = blockIdx.x * blockDim.x + threadIdx.x;
  if (i >= B_ * S_ * C_) return;
  int c = i % C_;
  int s = (i / C_) % S_;
  int b = i / (C_ * S_);
  int Lb = vid_len[b] * K_;
  int st = (s * Lb) / S_;
  int en = ((s + 1) * Lb + S_ - 1) / S_;
  float m = -3.4e38f;
  const float* p = outb + ((size_t)b * SL_) * C_ + c;
  for (int t = st; t < en; ++t) m = fmaxf(m, p[(size_t)t * C_]);
  flat[(size_t)b * (C_ * S_) + (size_t)c * S_ + s] = m;
}

// st/en heads: one wave per output scalar (2 * B * S waves)
__global__ void heads_kernel(const float* __restrict__ flat,
                             const float* __restrict__ w_st, const float* __restrict__ b_st,
                             const float* __restrict__ w_en, const float* __restrict__ b_en,
                             float* __restrict__ outp)
{
  int id = blockIdx.x;                     // 0 .. 2*B_*S_-1
  int lane = threadIdx.x;
  int which = id / (B_ * S_);
  int rem = id % (B_ * S_);
  int b = rem / S_, s = rem % S_;
  const float* w = (which ? w_en : w_st) + (size_t)s * (C_ * S_);
  const float* f = flat + (size_t)b * (C_ * S_);
  float acc = 0.f;
  for (int c = lane; c < C_ * S_; c += 32) acc += f[c] * w[c];
#pragma unroll
  for (int m = 16; m > 0; m >>= 1) acc += __shfl_xor(acc, m, 32);
  if (lane == 0)
    outp[which * (B_ * S_) + b * S_ + s] = acc + (which ? b_en[s] : b_st[s]);
}

// pred[b, j, nw] = dot(h[b,nw,:], w_p2[j,:]) + b_p2[j], one wave per scalar
__global__ void pred_kernel(const float* __restrict__ h,
                            const float* __restrict__ w_p2, const float* __restrict__ b_p2,
                            float* __restrict__ pred)
{
  int id = blockIdx.x;                     // 0 .. B_*NW_*2-1
  int lane = threadIdx.x;
  int j  = id % 2;
  int nw = (id / 2) % NW_;
  int b  = id / (2 * NW_);
  const float* hr = h + ((size_t)b * NW_ + nw) * C_;
  const float* w  = w_p2 + (size_t)j * C_;
  float acc = 0.f;
  for (int c = lane; c < C_; c += 32) acc += hr[c] * w[c];
#pragma unroll
  for (int m = 16; m > 0; m >>= 1) acc += __shfl_xor(acc, m, 32);
  if (lane == 0) pred[((size_t)b * 2 + j) * NW_ + nw] = acc + b_p2[j];
}

// ---------------------------------------------------------------------------
// host launcher
// ---------------------------------------------------------------------------
extern "C" void kernel_launch(void* const* d_in, const int* in_sizes, int n_in,
                              void* d_out, int out_size, void* d_ws, size_t ws_size,
                              hipStream_t stream)
{
  (void)in_sizes; (void)n_in; (void)out_size; (void)ws_size;

  const float* vis   = (const float*)d_in[0];
  const float* query = (const float*)d_in[1];
  const int*   vlen  = (const int*)d_in[2];
  const float* w_v1  = (const float*)d_in[3];  const float* b_v1 = (const float*)d_in[4];
  const float* w_v2  = (const float*)d_in[5];  const float* b_v2 = (const float*)d_in[6];
  const float* w_s1  = (const float*)d_in[7];  const float* b_s1 = (const float*)d_in[8];
  const float* w_s2  = (const float*)d_in[9];  const float* b_s2 = (const float*)d_in[10];
  const float* w_in  = (const float*)d_in[11]; const float* b_in = (const float*)d_in[12];
  const float* w_out = (const float*)d_in[13]; const float* b_out= (const float*)d_in[14];
  const float* w_p1  = (const float*)d_in[15]; const float* b_p1 = (const float*)d_in[16];
  const float* w_p2  = (const float*)d_in[17]; const float* b_p2 = (const float*)d_in[18];
  const float* w_st  = (const float*)d_in[19]; const float* b_st = (const float*)d_in[20];
  const float* w_en  = (const float*)d_in[21]; const float* b_en = (const float*)d_in[22];
  float* outp = (float*)d_out;

  char* ws = (char*)d_ws;
  size_t off = 0;
  auto alloc = [&](size_t bytes) -> char* {
    char* p = ws + off;
    off = (off + bytes + 255) & ~(size_t)255;
    return p;
  };
  float* vpe    = (float*)alloc((size_t)B_ * T_ * C_ * 4);
  float* vbuf   = (float*)alloc((size_t)B_ * T_ * C_ * 4);
  float* qpe    = (float*)alloc((size_t)B_ * LQ_ * QD_ * 4);
  float* e1     = (float*)alloc((size_t)B_ * LQ_ * C_ * 4);
  float* e2     = (float*)alloc((size_t)B_ * LQ_ * C_ * 4);
  int*   labels = (int*)  alloc((size_t)B_ * T_ * 4);
  int*   modeb  = (int*)  alloc((size_t)B_ * NW_ * K_ * 4);
  float* pp     = (float*)alloc((size_t)B_ * SL_ * C_ * 4);
  float* qq     = (float*)alloc((size_t)B_ * SL_ * C_ * 4);
  float* kk     = (float*)alloc((size_t)B_ * SL_ * C_ * 4);
  float* Qb     = (float*)alloc((size_t)B_ * SL_ * C_ * 4);
  float* Kb     = (float*)alloc((size_t)B_ * SL_ * C_ * 4);
  float* Vb     = (float*)alloc((size_t)B_ * SL_ * C_ * 4);
  float* sc     = (float*)alloc((size_t)B_ * H_ * SL_ * SLP_ * 4);
  float* ao     = (float*)alloc((size_t)B_ * SL_ * C_ * 4);
  float* ob     = (float*)alloc((size_t)B_ * SL_ * C_ * 4);
  float* hb     = (float*)alloc((size_t)B_ * NW_ * C_ * 4);
  float* flat   = (float*)alloc((size_t)B_ * C_ * S_ * 4);

  // 1. positional encodings
  { int tot = B_ * T_ * C_;
    pe_add_kernel<<<(tot + 255) / 256, 256, 0, stream>>>(vis, vpe, tot, T_, C_); }
  { int tot = B_ * LQ_ * QD_;
    pe_add_kernel<<<(tot + 255) / 256, 256, 0, stream>>>(query, qpe, tot, LQ_, QD_); }

  // 2. v = PE(vis) @ w_v1.T + b_v1        (2048 x 1024 x 1024)
  gemm_wmma_kernel<false, 0><<<dim3(C_ / 64, (B_ * T_ + 63) / 64), 128, 0, stream>>>(
      vpe, C_, w_v1, C_, b_v1, vbuf, C_, B_ * T_, C_, C_, C_);

  // 3. e1 / e2 = PE(q) @ w_s{1,2}.T       (192 x 1024 x 768)
  gemm_wmma_kernel<false, 0><<<dim3(C_ / 64, (B_ * LQ_ + 63) / 64), 128, 0, stream>>>(
      qpe, QD_, w_s1, QD_, b_s1, e1, C_, B_ * LQ_, C_, QD_, QD_);
  gemm_wmma_kernel<false, 0><<<dim3(C_ / 64, (B_ * LQ_ + 63) / 64), 128, 0, stream>>>(
      qpe, QD_, w_s2, QD_, b_s2, e2, C_, B_ * LQ_, C_, QD_, QD_);

  // 4. labels = argmax_l <v, e1>
  sim_argmax_kernel<<<B_ * T_, 32, 0, stream>>>(vbuf, e1, labels);

  // 5. per-chunk label mode, then gather kk = e2[mode]
  { int tot = B_ * NW_ * K_;
    mode_kernel<<<(tot + 255) / 256, 256, 0, stream>>>(labels, modeb); }
  { size_t tot = (size_t)B_ * SL_ * C_;
    gather_modk_kernel<<<(unsigned)((tot + 255) / 256), 256, 0, stream>>>(e2, modeb, kk); }

  // 6. pooled = mean-window(v) @ w_v2.T   (7712 x 1024 x 1024)
  { size_t tot = (size_t)B_ * SL_ * C_;
    pool_mean_kernel<<<(unsigned)((tot + 255) / 256), 256, 0, stream>>>(vbuf, pp); }
  gemm_wmma_kernel<false, 0><<<dim3(C_ / 64, (B_ * SL_ + 63) / 64), 128, 0, stream>>>(
      pp, C_, w_v2, C_, b_v2, qq, C_, B_ * SL_, C_, C_, C_);

  // 7. Q / K / V projections (w_in split along rows)
  gemm_wmma_kernel<false, 0><<<dim3(C_ / 64, (B_ * SL_ + 63) / 64), 128, 0, stream>>>(
      qq, C_, w_in, C_, b_in, Qb, C_, B_ * SL_, C_, C_, C_);
  gemm_wmma_kernel<false, 0><<<dim3(C_ / 64, (B_ * SL_ + 63) / 64), 128, 0, stream>>>(
      kk, C_, w_in + (size_t)C_ * C_, C_, b_in + C_, Kb, C_, B_ * SL_, C_, C_, C_);
  gemm_wmma_kernel<false, 0><<<dim3(C_ / 64, (B_ * SL_ + 63) / 64), 128, 0, stream>>>(
      qq, C_, w_in + 2 * (size_t)C_ * C_, C_, b_in + 2 * C_, Vb, C_, B_ * SL_, C_, C_, C_);

  // 8. attention per (b,h): scores, softmax, attn @ V
  for (int b = 0; b < B_; ++b)
    for (int h = 0; h < H_; ++h) {
      const float* Ab = Qb + ((size_t)b * SL_) * C_ + (size_t)h * HD_;
      const float* Bb = Kb + ((size_t)b * SL_) * C_ + (size_t)h * HD_;
      float* Sb = sc + ((size_t)(b * H_ + h)) * SL_ * SLP_;
      gemm_wmma_kernel<false, 0><<<dim3((SL_ + 63) / 64, (SL_ + 63) / 64), 128, 0, stream>>>(
          Ab, C_, Bb, C_, nullptr, Sb, SLP_, SL_, SL_, HD_, HD_);
    }
  softmax_kernel<<<B_ * H_ * SL_, 256, 0, stream>>>(sc);
  for (int b = 0; b < B_; ++b)
    for (int h = 0; h < H_; ++h) {
      const float* Pb = sc + ((size_t)(b * H_ + h)) * SL_ * SLP_;
      const float* Vh = Vb + ((size_t)b * SL_) * C_ + (size_t)h * HD_;
      float* Ob = ao + ((size_t)b * SL_) * C_ + (size_t)h * HD_;
      gemm_wmma_kernel<true, 0><<<dim3(HD_ / 64, (SL_ + 63) / 64), 128, 0, stream>>>(
          Pb, SLP_, Vh, C_, nullptr, Ob, C_, SL_, HD_, SLP_, SL_);
    }

  // 9. out projection
  gemm_wmma_kernel<false, 0><<<dim3(C_ / 64, (B_ * SL_ + 63) / 64), 128, 0, stream>>>(
      ao, C_, w_out, C_, b_out, ob, C_, B_ * SL_, C_, C_, C_);

  // 10. h = relu(out.reshape(B*NW, 4C) @ w_p1.T)  (1928 x 1024 x 4096)
  gemm_wmma_kernel<false, 1><<<dim3(C_ / 64, (B_ * NW_ + 63) / 64), 128, 0, stream>>>(
      ob, K_ * C_, w_p1, K_ * C_, b_p1, hb, C_, B_ * NW_, C_, K_ * C_, K_ * C_);

  // 11. pred -> d_out[0 .. B*2*NW)
  pred_kernel<<<B_ * NW_ * 2, 32, 0, stream>>>(hb, w_p2, b_p2, outp);

  // 12. segment max-pool + st/en heads -> d_out tail
  { int tot = B_ * S_ * C_;
    segpool_kernel<<<(tot + 255) / 256, 256, 0, stream>>>(ob, vlen, flat); }
  heads_kernel<<<2 * B_ * S_, 32, 0, stream>>>(flat, w_st, b_st, w_en, b_en,
                                               outp + (size_t)B_ * 2 * NW_);
}
